// SWMSA_7318624272452
// MI455X (gfx1250) — matrix operand order repaired and checked by reading, not compiled, not run on hardware
//
#include <hip/hip_runtime.h>
#include <hip/hip_bf16.h>

typedef _Float16 v16h __attribute__((ext_vector_type(16)));
typedef _Float16 h8v  __attribute__((ext_vector_type(8)));
typedef _Float16 h4v  __attribute__((ext_vector_type(4)));
typedef float    v8f  __attribute__((ext_vector_type(8)));
typedef float    f4v  __attribute__((ext_vector_type(4)));

#define HH   56
#define WW   56
#define WSZ  7
#define SSH  3
#define NHD  8
#define EE   256
#define HD   32
#define NWIN 64
#define WD   49

#define XST  264            // Xs row stride in halves (256 + 8 pad)
#define QST  40             // Q/K row stride (32 + 8 pad)
#define VST  72             // Vt row stride (64 + 8 pad)
#define AST  72             // attn row stride (64 + 8 pad)
#define HSCR (2*64*QST + 32*VST)   // per-head scratch = 7424 halves

// ---- workspace layout (bytes) ----
#define OFF_PWB   0          // proj_w fragment-major f16: 8*16 frags * 32 lanes * 16 halves = 131072 B
#define OFF_QKVB  131072     // qkv_w fragment-major f16 : 6 frags  * 32 * 16 halves       = 6144 B
#define OFF_BIAS  137216     // bias  C-frag-major  f32  : 8*16 frags * 32 lanes * 8        = 131072 B
#define OFF_REG   268288     // region ids int32         : 64*49                            = 12544 B
#define N_PWB  65536
#define N_QKVB 3072
#define N_BIAS 32768
#define N_REG  3136
#define N_PREP (N_PWB + N_QKVB + N_BIAS + N_REG)

union V16U { v16h v; h8v h[2]; };

static __device__ __forceinline__ v8f wmma16(v16h a, v16h b, v8f c) {
  return __builtin_amdgcn_wmma_f32_16x16x32_f16(false, a, false, b, (short)0, c, false, false);
}

// A-matrix 16x32 fragment from row-major [rows][stride] f16 LDS.
// lane row = row0+l16 ; halves e -> K = col0 + half*16 + e  (contiguous)
static __device__ __forceinline__ v16h load_a(const _Float16* base, int stride,
                                              int row0, int col0, int l16, int half) {
  const _Float16* p = base + (row0 + l16) * stride + col0 + half * 16;
  V16U u;
  u.h[0] = *(const h8v*)p;
  u.h[1] = *(const h8v*)(p + 8);
  return u.v;
}

// B-matrix 32x16 fragment, from *transposed* storage [N][K] row-major f16 LDS.
// lane column N = n0+l16 ; halves e -> K = k0 + half*16 + e  (contiguous)
static __device__ __forceinline__ v16h load_bt(const _Float16* base, int stride,
                                               int n0, int k0, int l16, int half) {
  const _Float16* p = base + (n0 + l16) * stride + k0 + half * 16;
  V16U u;
  u.h[0] = *(const h8v*)p;
  u.h[1] = *(const h8v*)(p + 8);
  return u.v;
}

// ---------------- prep: swizzle weights into WMMA-fragment-major layouts ----------------
__global__ void swmsa_prep(const float* __restrict__ qkv_w, const float* __restrict__ proj_w,
                           const float* __restrict__ rpb, _Float16* __restrict__ pwB,
                           _Float16* __restrict__ qkvB, float* __restrict__ biasF,
                           int* __restrict__ region) {
  const int g = blockIdx.x * 256 + threadIdx.x;
  if (g < N_PWB) {                       // proj_w B-fragments: frag f = kt*16+nt
    const int e = g & 15, ln = (g >> 4) & 31, f = g >> 9;
    const int nt = f & 15, kt = f >> 4;
    const int K = kt * 32 + (ln >> 4) * 16 + e;
    const int N = nt * 16 + (ln & 15);
    pwB[g] = (_Float16)proj_w[K * EE + N];
  } else if (g < N_PWB + N_QKVB) {       // qkv_w B-fragments (K=32, 6 N-tiles)
    const int i = g - N_PWB;
    const int e = i & 15, ln = (i >> 4) & 31, nt = i >> 9;
    const int K = (ln >> 4) * 16 + e;
    const int N = nt * 16 + (ln & 15);
    qkvB[i] = (_Float16)qkv_w[K * 96 + N];
  } else if (g < N_PWB + N_QKVB + N_BIAS) {  // rel-pos bias in C-fragment layout, pad baked in
    const int i = g - (N_PWB + N_QKVB);
    const int r = i & 7, ln = (i >> 3) & 31, f = i >> 8;  // f = h*16 + mt*4 + nt
    const int nt = f & 3, mt = (f >> 2) & 3, h = f >> 4;
    const int M = mt * 16 + (ln >> 4) * 8 + r;
    const int N = nt * 16 + (ln & 15);
    float v;
    if (N >= WD)      v = -1e30f;        // mask padded key columns
    else if (M >= WD) v = 0.f;
    else {
      const int yi = M / WSZ, xi = M - yi * WSZ;
      const int yj = N / WSZ, xj = N - yj * WSZ;
      const int rel = (yi - yj + WSZ - 1) * (2 * WSZ - 1) + (xi - xj + WSZ - 1);
      v = rpb[rel * NHD + h];
    }
    biasF[i] = v;
  } else if (g < N_PREP) {               // shifted-window region ids (for the -100 mask)
    const int i = g - (N_PWB + N_QKVB + N_BIAS);
    const int w = i / WD, t = i - w * WD;
    const int wr = w >> 3, wc = w & 7;
    const int r = t / WSZ, c = t - r * WSZ;
    int hh = wr * WSZ + r + SSH; if (hh >= HH) hh -= HH;
    int ww = wc * WSZ + c + SSH; if (ww >= WW) ww -= WW;
    const int ch = hh < 49 ? 0 : (hh < 53 ? 1 : 2);
    const int cw = ww < 49 ? 0 : (ww < 53 ? 1 : 2);
    region[i] = ch * 3 + cw;
  }
}

// ---------------- fused W-MSA: one workgroup = one (batch, window); wave = head ----------------
__global__ __launch_bounds__(256) void swmsa_main(
    const float* __restrict__ x, const float* __restrict__ qkv_b,
    const float* __restrict__ proj_b, const _Float16* __restrict__ pwB,
    const _Float16* __restrict__ qkvBg, const float* __restrict__ biasF,
    const int* __restrict__ region, float* __restrict__ out) {
  __shared__ _Float16 Xs[64 * XST];        // window activations (f16), later reused as O
  __shared__ _Float16 HS[NHD * HSCR];      // per-head Q | K | Vt scratch (attn overlays Q+K)
  __shared__ _Float16 qkvBs[6 * 32 * 16];
  __shared__ float    qbs[96];
  __shared__ int      rgs[64];

  const int tid  = threadIdx.x;
  const int lane = tid & 31;
  const int wv   = tid >> 5;
  const int half = lane >> 4;
  const int l16  = lane & 15;
  const int b  = blockIdx.x >> 6;
  const int w  = blockIdx.x & 63;
  const int wr = w >> 3, wc = w & 7;

  for (int i = tid; i < 6 * 32 * 16; i += 256) qkvBs[i] = qkvBg[i];
  if (tid < 96) qbs[tid] = qkv_b[tid];
  if (tid < WD) rgs[tid] = region[w * WD + tid];

  // ---- S1: gather cyclically-shifted window -> Xs (f16), rows 49..63 zero-padded
  for (int i = tid; i < 64 * 64; i += 256) {
    const int t = i >> 6, e4 = (i & 63) << 2;
    h4v hv = {(_Float16)0.f, (_Float16)0.f, (_Float16)0.f, (_Float16)0.f};
    if (t < WD) {
      const int r = t / WSZ, c = t - r * WSZ;
      int hh = wr * WSZ + r + SSH; if (hh >= HH) hh -= HH;
      int ww = wc * WSZ + c + SSH; if (ww >= WW) ww -= WW;
      const f4v f = *(const f4v*)&x[(((size_t)b * HH * WW) + hh * WW + ww) * EE + e4];
      hv.x = (_Float16)f.x; hv.y = (_Float16)f.y; hv.z = (_Float16)f.z; hv.w = (_Float16)f.w;
    }
    *(h4v*)&Xs[t * XST + e4] = hv;
  }
  __syncthreads();

  const int h = wv;                       // this wave's head
  _Float16* Qs = HS + h * HSCR;           // [64][QST]  (Q, pre-scaled by 1/sqrt(HD))
  _Float16* Ks = Qs + 64 * QST;           // [64][QST]  (K, stored [token][hd])
  _Float16* Vt = Qs + 2 * 64 * QST;       // [32][VST]  (V transposed: [hd][token])
  _Float16* At = Qs;                      // [64][AST]  attn overlays Q+K (safe: frags hoisted)

  // ---- S2: per-head QKV = X_h(49x32) @ qkv_w(32x96) + qkv_b
  v16h xa[4];
  #pragma unroll
  for (int mt = 0; mt < 4; ++mt) xa[mt] = load_a(Xs, XST, mt * 16, h * HD, l16, half);
  #pragma unroll
  for (int nt = 0; nt < 6; ++nt) {
    const v16h bf = *(const v16h*)&qkvBs[(nt * 32 + lane) * 16];
    const int n = nt * 16 + l16;
    const float qb = qbs[n];
    #pragma unroll
    for (int mt = 0; mt < 4; ++mt) {
      v8f z = {0.f, 0.f, 0.f, 0.f, 0.f, 0.f, 0.f, 0.f};
      const v8f acc = wmma16(xa[mt], bf, z);
      #pragma unroll
      for (int r = 0; r < 8; ++r) {
        const int m = mt * 16 + half * 8 + r;
        const float v = acc[r] + qb;
        if (n < HD)            Qs[m * QST + n] = (_Float16)(v * 0.1767766953f); // 1/sqrt(32)
        else if (n < 2 * HD)   Ks[m * QST + (n - HD)] = (_Float16)v;
        else                   Vt[(n - 2 * HD) * VST + m] = (_Float16)v;
      }
    }
  }

  // ---- S3: scores = Q @ K^T (+bias +shift-mask), row softmax in registers
  v16h qf[4], kf[4];
  #pragma unroll
  for (int mt = 0; mt < 4; ++mt) qf[mt] = load_a(Qs, QST, mt * 16, 0, l16, half);
  #pragma unroll
  for (int nt = 0; nt < 4; ++nt) kf[nt] = load_bt(Ks, QST, nt * 16, 0, l16, half);

  #pragma unroll
  for (int mt = 0; mt < 4; ++mt) {
    v8f sc[4];
    #pragma unroll
    for (int nt = 0; nt < 4; ++nt) {
      v8f z = {0.f, 0.f, 0.f, 0.f, 0.f, 0.f, 0.f, 0.f};
      sc[nt] = wmma16(qf[mt], kf[nt], z);
      const float* bp = biasF + ((((h * 16 + mt * 4 + nt) * 32) + lane) << 3);
      const int n = nt * 16 + l16;
      const int rn = (n < WD) ? rgs[n] : 0;
      #pragma unroll
      for (int r = 0; r < 8; ++r) {
        const int m = mt * 16 + half * 8 + r;
        float v = sc[nt][r] + bp[r];
        if (n < WD && m < WD && rgs[m] != rn) v -= 100.f;
        sc[nt][r] = v;
      }
    }
    #pragma unroll
    for (int r = 0; r < 8; ++r) {   // softmax over the 64-wide row (16 lanes x 4 tiles)
      float mx = fmaxf(fmaxf(sc[0][r], sc[1][r]), fmaxf(sc[2][r], sc[3][r]));
      #pragma unroll
      for (int o = 1; o < 16; o <<= 1) mx = fmaxf(mx, __shfl_xor(mx, o, 32));
      float sum = 0.f;
      #pragma unroll
      for (int nt = 0; nt < 4; ++nt) { const float ev = __expf(sc[nt][r] - mx); sc[nt][r] = ev; sum += ev; }
      #pragma unroll
      for (int o = 1; o < 16; o <<= 1) sum += __shfl_xor(sum, o, 32);
      const float inv = 1.f / sum;
      #pragma unroll
      for (int nt = 0; nt < 4; ++nt) sc[nt][r] *= inv;
    }
    #pragma unroll
    for (int nt = 0; nt < 4; ++nt) {
      const int n = nt * 16 + l16;
      #pragma unroll
      for (int r = 0; r < 8; ++r) {
        const int m = mt * 16 + half * 8 + r;
        At[m * AST + n] = (_Float16)sc[nt][r];
      }
    }
  }

  // ---- S4: O_h = attn(64x64) @ V(64x32) -> Xs columns [h*32, h*32+32)
  #pragma unroll
  for (int vt = 0; vt < 2; ++vt) {
    const v16h b0 = load_bt(Vt, VST, vt * 16, 0,  l16, half);
    const v16h b1 = load_bt(Vt, VST, vt * 16, 32, l16, half);
    #pragma unroll
    for (int mt = 0; mt < 4; ++mt) {
      const v16h a0 = load_a(At, AST, mt * 16, 0,  l16, half);
      const v16h a1 = load_a(At, AST, mt * 16, 32, l16, half);
      v8f z = {0.f, 0.f, 0.f, 0.f, 0.f, 0.f, 0.f, 0.f};
      v8f acc = wmma16(a0, b0, z);
      acc = wmma16(a1, b1, acc);
      const int n = h * HD + vt * 16 + l16;
      #pragma unroll
      for (int r = 0; r < 8; ++r) {
        const int m = mt * 16 + half * 8 + r;
        Xs[m * XST + n] = (_Float16)acc[r];
      }
    }
  }
  __syncthreads();

  // ---- S5: out = O(49x256) @ proj_w(256x256) + proj_b ; wave handles 2 N-tiles
  #pragma unroll
  for (int ni = 0; ni < 2; ++ni) {
    const int nt = wv * 2 + ni;
    v8f acc[4];
    #pragma unroll
    for (int mt = 0; mt < 4; ++mt) acc[mt] = (v8f){0.f, 0.f, 0.f, 0.f, 0.f, 0.f, 0.f, 0.f};
    #pragma unroll
    for (int kt = 0; kt < 8; ++kt) {
      const v16h bf = *(const v16h*)&pwB[(((kt * 16 + nt) * 32) + lane) * 16];  // L2-resident
      #pragma unroll
      for (int mt = 0; mt < 4; ++mt) {
        const v16h a = load_a(Xs, XST, mt * 16, kt * 32, l16, half);
        acc[mt] = wmma16(a, bf, acc[mt]);
      }
    }
    const int n = nt * 16 + l16;
    const float pb = proj_b[n];
    #pragma unroll
    for (int mt = 0; mt < 4; ++mt) {
      #pragma unroll
      for (int r = 0; r < 8; ++r) {
        const int m = mt * 16 + half * 8 + r;
        if (m < WD) {
          const int rr = m / WSZ, cc = m - rr * WSZ;
          int hh = wr * WSZ + rr + SSH; if (hh >= HH) hh -= HH;
          int ww = wc * WSZ + cc + SSH; if (ww >= WW) ww -= WW;
          out[(((size_t)b * HH * WW) + hh * WW + ww) * EE + n] = acc[mt][r] + pb;
        }
      }
    }
  }
}

extern "C" void kernel_launch(void* const* d_in, const int* in_sizes, int n_in,
                              void* d_out, int out_size, void* d_ws, size_t ws_size,
                              hipStream_t stream) {
  const float* x      = (const float*)d_in[0];
  const float* qkv_w  = (const float*)d_in[1];
  const float* qkv_b  = (const float*)d_in[2];
  const float* proj_w = (const float*)d_in[3];
  const float* proj_b = (const float*)d_in[4];
  const float* rpb    = (const float*)d_in[5];
  float* out = (float*)d_out;

  char* ws = (char*)d_ws;
  _Float16* pwB    = (_Float16*)(ws + OFF_PWB);
  _Float16* qkvB   = (_Float16*)(ws + OFF_QKVB);
  float*    biasF  = (float*)(ws + OFF_BIAS);
  int*      region = (int*)(ws + OFF_REG);

  swmsa_prep<<<(N_PREP + 255) / 256, 256, 0, stream>>>(qkv_w, proj_w, rpb, pwB, qkvB, biasF, region);
  swmsa_main<<<32 * NWIN, 256, 0, stream>>>(x, qkv_b, proj_b, pwB, qkvB, biasF, region, out);
}
